// ICMCFMDecoder_29575144801160
// MI455X (gfx1250) — matrix operand order repaired and checked by reading, not compiled, not run on hardware
//
#include <hip/hip_runtime.h>
#include <hip/hip_bf16.h>
#include <math.h>

typedef __bf16 bf16;
typedef __bf16 v16bf  __attribute__((ext_vector_type(16)));
typedef float  v8f    __attribute__((ext_vector_type(8)));
typedef __bf16 bf16x4 __attribute__((ext_vector_type(4)));

#define N_ROWS 65536
#define H_DIM  1024
#define KCH1   33            // ceil(1027/32) with zero padding -> 1056
#define KCH2   32
#define NSUB_TOT 64          // H/16 output sub-tiles per row of W

__device__ __forceinline__ bf16 f2bf(float f) {
  unsigned u = __builtin_bit_cast(unsigned, f);
  unsigned r = (u + 0x7fffu + ((u >> 16) & 1u)) >> 16;
  return __builtin_bit_cast(bf16, (unsigned short)r);
}

__device__ __forceinline__ float siluf(float v) {
  return v / (1.f + __expf(-v));
}

// One wave-wide async copy: 16B per lane, per-lane LDS destination.
__device__ __forceinline__ void async_b128(unsigned lds_addr, const void* gsrc) {
  asm volatile("global_load_async_to_lds_b128 %0, %1, off"
               :: "v"(lds_addr), "v"((unsigned long long)(size_t)gsrc)
               : "memory");
}
__device__ __forceinline__ unsigned lds_u32(const void* p) {
  return (unsigned)(size_t)p;
}

// ---------------------------------------------------------------- init
__global__ void k_init(float* __restrict__ pred_acc, float* __restrict__ scal) {
  int i = blockIdx.x * 256 + threadIdx.x;
  if (i < N_ROWS) pred_acc[i] = 0.f;
  if (i < 4) scal[i] = 0.f;
}

// --------------------------------------------------- row prep (GEMVs + cvt)
__global__ void __launch_bounds__(256)
k_row_prep(const float* __restrict__ h, const float* __restrict__ targets,
           const float* __restrict__ x0, const float* __restrict__ t,
           const float* __restrict__ Ws, const float* __restrict__ bs,
           const float* __restrict__ Wc, const float* __restrict__ bc,
           bf16* __restrict__ hb, bf16* __restrict__ extras,
           float* __restrict__ scal) {
  int wave = threadIdx.x >> 5;
  int lane = threadIdx.x & 31;
  int row  = blockIdx.x * 8 + wave;
  const float* hr = h + (size_t)row * H_DIM;

  float ds = 0.f, dc = 0.f;
#pragma unroll
  for (int j = 0; j < 8; ++j) {
    int k = j * 128 + lane * 4;
    float4 hv = *(const float4*)(hr + k);
    float4 wsv = *(const float4*)(Ws + k);
    float4 wcv = *(const float4*)(Wc + k);
    ds += hv.x * wsv.x + hv.y * wsv.y + hv.z * wsv.z + hv.w * wsv.w;
    dc += hv.x * wcv.x + hv.y * wcv.y + hv.z * wcv.z + hv.w * wcv.w;
    bf16x4 b = { f2bf(hv.x), f2bf(hv.y), f2bf(hv.z), f2bf(hv.w) };
    *(bf16x4*)(hb + (size_t)row * H_DIM + k) = b;
  }
#pragma unroll
  for (int off = 16; off > 0; off >>= 1) {
    ds += __shfl_xor(ds, off, 32);
    dc += __shfl_xor(dc, off, 32);
  }

  __shared__ float red[3][8];
  if (lane == 0) {
    float z = ds + bs[0];
    float cpre = dc + z * Wc[H_DIM] + bc[0];
    float sp = (cpre > 0.f) ? (cpre + log1pf(__expf(-cpre))) : log1pf(__expf(cpre));
    float clam = sp + 1e-6f;
    float tg = targets[row];
    float y01 = (tg > 0.f) ? 1.f : 0.f;
    float sterm = fmaxf(z, 0.f) - z * y01 + log1pf(__expf(-fabsf(z)));
    float ctgt = log1pf(fmaxf(tg, 0.f));
    float cterm = (clam - ctgt) * (clam - ctgt);
    red[0][wave] = sterm;
    red[1][wave] = cterm;
    red[2][wave] = y01;
    float tt = t[row];
    float xt = (1.f - tt) * x0[row] + tt * ctgt;
    bf16 z0 = f2bf(0.f);
    bf16x4 e0 = { f2bf(clam), f2bf(xt), f2bf(tt), z0 };
    bf16x4 ez = { z0, z0, z0, z0 };
    bf16x4* ep = (bf16x4*)(extras + (size_t)row * 32);
    ep[0] = e0;
#pragma unroll
    for (int j = 1; j < 8; ++j) ep[j] = ez;
  }
  __syncthreads();
  if (threadIdx.x == 0) {
    float a = 0.f, b2 = 0.f, c = 0.f;
    for (int w = 0; w < 8; ++w) { a += red[0][w]; b2 += red[1][w]; c += red[2][w]; }
    atomicAdd(&scal[0], a);
    atomicAdd(&scal[1], b2);
    atomicAdd(&scal[2], c);
  }
}

// ------------------------------------------- weight convert + frag swizzle
// Output: 1KB blocks indexed [kc*64 + nsub]; block byte = lane*32 + slot*2
// B layout (32x16 KxN): lane = (n&15) + ((k>>3)&1)*16 ; slot = (k&7) + 8*((k>>4)&1)
__global__ void k_wprep(const float* __restrict__ W, int Krows,
                        bf16* __restrict__ Wsw) {
  int blk = blockIdx.x;           // kc * NSUB_TOT + nsub
  int kc = blk / NSUB_TOT;
  int nsub = blk % NSUB_TOT;
  bf16* dst = Wsw + (size_t)blk * 512;
  for (int e = threadIdx.x; e < 512; e += 256) {
    int lanei = e >> 4;
    int slot  = e & 15;
    int k = kc * 32 + (slot & 7) + 8 * (lanei >> 4) + 16 * (slot >> 3);
    int n = nsub * 16 + (lanei & 15);
    float v = (k < Krows) ? W[(size_t)k * H_DIM + n] : 0.f;
    dst[lanei * 16 + slot] = f2bf(v);
  }
}

// ------------------------------------------------------------- GEMM core
// 128x128 macro tile; 8 waves = 4(M) x 2(N); wave tile 32x64 = 2x4 subtiles.
// A fragment layout (16x32 MxK): lane = (m&15) + ((k>>3)&1)*16 ;
//                                slot = (k&7) + 8*((k>>4)&1)
// Staging uses double-buffered GLOBAL_LOAD_ASYNC_TO_LDS_B128 (4 async
// instructions per wave per chunk) + s_wait_asynccnt pipelining.
__device__ __forceinline__ void stage_async(
    const bf16* __restrict__ A, const bf16* __restrict__ extras,
    const bf16* __restrict__ Wsw, int kc, int r0, int cb,
    bf16* Ab, bf16* Bb, int tid) {
  // A: 512 x 16B segments; seg s -> (m = s>>2, kg = s&3); 16B row-major
  // segment lands in 16 contiguous bytes of the fragment block.
#pragma unroll
  for (int si = 0; si < 2; ++si) {
    int s = tid * 2 + si;
    int m = s >> 2, kg = s & 3;
    const bf16* src;
    if (extras != nullptr && kc == KCH1 - 1)
      src = extras + (size_t)(r0 + m) * 32 + kg * 8;
    else
      src = A + (size_t)(r0 + m) * H_DIM + (size_t)kc * 32 + kg * 8;
    int lanei = (m & 15) + (kg & 1) * 16;
    unsigned dst = lds_u32((char*)Ab + (m >> 4) * 1024 + lanei * 32 + (kg >> 1) * 16);
    async_b128(dst, src);
  }
  // B: pre-swizzled, contiguous 8KB
  const char* bsrc = (const char*)(Wsw + ((size_t)kc * NSUB_TOT + (cb >> 4)) * 512);
  unsigned bdst = lds_u32(Bb);
  async_b128(bdst + tid * 16u, bsrc + tid * 16);
  async_b128(bdst + 4096u + tid * 16u, bsrc + 4096 + tid * 16);
}

__device__ __forceinline__ void gemm_tile(
    const bf16* __restrict__ A, const bf16* __restrict__ extras,
    const bf16* __restrict__ Wsw, int kchunks, int r0, int cb,
    v8f acc[2][4]) {
  __shared__ __align__(16) bf16 Abuf[2][128 * 32];
  __shared__ __align__(16) bf16 Bbuf[2][32 * 128];

  int tid = threadIdx.x;
  int wave = tid >> 5, lane = tid & 31;
  int waveM = wave >> 1, waveN = wave & 1;

  stage_async(A, extras, Wsw, 0, r0, cb, Abuf[0], Bbuf[0], tid);

  for (int kc = 0; kc < kchunks; ++kc) {
    int cur = kc & 1;
    if (kc + 1 < kchunks) {
      stage_async(A, extras, Wsw, kc + 1, r0, cb, Abuf[cur ^ 1], Bbuf[cur ^ 1], tid);
      asm volatile("s_wait_asynccnt 0x4" ::: "memory");  // chunk kc resident
    } else {
      asm volatile("s_wait_asynccnt 0x0" ::: "memory");
    }
    __syncthreads();

    v16bf af[2], bfr[4];
#pragma unroll
    for (int mi = 0; mi < 2; ++mi) {
      int msub = waveM * 2 + mi;
      af[mi] = *(const v16bf*)((const char*)Abuf[cur] + msub * 1024 + lane * 32);
    }
#pragma unroll
    for (int ni = 0; ni < 4; ++ni) {
      int nsub = waveN * 4 + ni;
      bfr[ni] = *(const v16bf*)((const char*)Bbuf[cur] + nsub * 1024 + lane * 32);
    }
#pragma unroll
    for (int mi = 0; mi < 2; ++mi)
#pragma unroll
      for (int ni = 0; ni < 4; ++ni)
        acc[mi][ni] = __builtin_amdgcn_wmma_f32_16x16x32_bf16(
            false, af[mi], false, bfr[ni], (short)0, acc[mi][ni], false, false);

    __syncthreads();  // protect buffer reuse at kc+2
  }
}

// ------------------------------------------------------ GEMM1: z1 = silu(.)
// grid = (colblocks, rowblocks): column-block fastest so 8 consecutive WGs
// reuse one A row-block out of L2.
__global__ void __launch_bounds__(256)
k_gemm1(const bf16* __restrict__ hb, const bf16* __restrict__ extras,
        const bf16* __restrict__ W1s, const float* __restrict__ b1,
        bf16* __restrict__ z1b) {
  int cb = blockIdx.x * 128, r0 = blockIdx.y * 128;
  v8f acc[2][4] = {};
  gemm_tile(hb, extras, W1s, KCH1, r0, cb, acc);

  int wave = threadIdx.x >> 5, lane = threadIdx.x & 31;
  int waveM = wave >> 1, waveN = wave & 1;
  int rhi = (lane >> 4) * 8;
  int nlo = lane & 15;
#pragma unroll
  for (int mi = 0; mi < 2; ++mi) {
#pragma unroll
    for (int ni = 0; ni < 4; ++ni) {
      int col = cb + (waveN * 4 + ni) * 16 + nlo;
      float bias = b1[col];
#pragma unroll
      for (int r = 0; r < 8; ++r) {
        int row = r0 + (waveM * 2 + mi) * 16 + r + rhi;
        float v = acc[mi][ni][r] + bias;
        z1b[(size_t)row * H_DIM + col] = f2bf(siluf(v));
      }
    }
  }
}

// ------------------------------- GEMM2: z2 = silu(.), fused pred = z2 @ W3
__global__ void __launch_bounds__(256)
k_gemm2(const bf16* __restrict__ z1b, const bf16* __restrict__ W2s,
        const float* __restrict__ b2, const float* __restrict__ W3,
        float* __restrict__ pred_acc) {
  int cb = blockIdx.x * 128, r0 = blockIdx.y * 128;
  v8f acc[2][4] = {};
  gemm_tile(z1b, nullptr, W2s, KCH2, r0, cb, acc);

  int wave = threadIdx.x >> 5, lane = threadIdx.x & 31;
  int waveM = wave >> 1, waveN = wave & 1;
  int rhi = (lane >> 4) * 8;
  int nlo = lane & 15;

  float rp[2][8];
#pragma unroll
  for (int mi = 0; mi < 2; ++mi)
#pragma unroll
    for (int r = 0; r < 8; ++r) rp[mi][r] = 0.f;

#pragma unroll
  for (int mi = 0; mi < 2; ++mi) {
#pragma unroll
    for (int ni = 0; ni < 4; ++ni) {
      int col = cb + (waveN * 4 + ni) * 16 + nlo;
      float bias = b2[col];
      float w3 = W3[col];
#pragma unroll
      for (int r = 0; r < 8; ++r) {
        float v = acc[mi][ni][r] + bias;
        rp[mi][r] += siluf(v) * w3;
      }
    }
  }
#pragma unroll
  for (int mi = 0; mi < 2; ++mi) {
#pragma unroll
    for (int r = 0; r < 8; ++r) {
      float v = rp[mi][r];
      v += __shfl_xor(v, 1, 32);
      v += __shfl_xor(v, 2, 32);
      v += __shfl_xor(v, 4, 32);
      v += __shfl_xor(v, 8, 32);
      if ((lane & 15) == 0) {
        int row = r0 + (waveM * 2 + mi) * 16 + r + rhi;
        atomicAdd(&pred_acc[row], v);
      }
    }
  }
}

// ----------------------------------------------------------- final losses
__global__ void __launch_bounds__(256)
k_fin_rows(const float* __restrict__ pred_acc, const float* __restrict__ targets,
           const float* __restrict__ x0, const float* __restrict__ b3,
           float* __restrict__ scal) {
  int i = blockIdx.x * 256 + threadIdx.x;
  int wave = threadIdx.x >> 5, lane = threadIdx.x & 31;
  float tg = targets[i];
  float mask = (tg > 0.f) ? 1.f : 0.f;
  float y = log1pf(fmaxf(tg, 0.f));
  float tv = y - x0[i];
  float pred = pred_acc[i] + b3[0];
  float d = pred - tv;
  float val = mask * d * d;
#pragma unroll
  for (int off = 16; off > 0; off >>= 1) val += __shfl_xor(val, off, 32);
  __shared__ float red[8];
  if (lane == 0) red[wave] = val;
  __syncthreads();
  if (threadIdx.x == 0) {
    float s = 0.f;
    for (int w = 0; w < 8; ++w) s += red[w];
    atomicAdd(&scal[3], s);
  }
}

__global__ void k_fin_out(const float* __restrict__ scal, float* __restrict__ out) {
  if (threadIdx.x == 0 && blockIdx.x == 0) {
    float s_loss = scal[0] / (float)N_ROWS;
    float c_loss = scal[1] / (float)N_ROWS;
    float npos = fmaxf(scal[2], 1.f);
    float main_loss = scal[3] / npos;
    out[0] = main_loss;
    out[1] = s_loss;
    out[2] = c_loss;
    out[3] = main_loss + 0.05f * s_loss + 0.05f * c_loss;
  }
}

// ------------------------------------------------------------------ launch
extern "C" void kernel_launch(void* const* d_in, const int* in_sizes, int n_in,
                              void* d_out, int out_size, void* d_ws, size_t ws_size,
                              hipStream_t stream) {
  (void)in_sizes; (void)n_in; (void)out_size; (void)ws_size;
  const float* h   = (const float*)d_in[0];
  const float* tg  = (const float*)d_in[1];
  const float* x0  = (const float*)d_in[2];
  const float* t   = (const float*)d_in[3];
  const float* Ws  = (const float*)d_in[4];
  const float* bs  = (const float*)d_in[5];
  const float* Wc  = (const float*)d_in[6];
  const float* bc  = (const float*)d_in[7];
  const float* W1  = (const float*)d_in[8];
  const float* b1  = (const float*)d_in[9];
  const float* W2  = (const float*)d_in[10];
  const float* b2  = (const float*)d_in[11];
  const float* W3  = (const float*)d_in[12];
  const float* b3  = (const float*)d_in[13];
  float* out = (float*)d_out;

  char* ws = (char*)d_ws;
  size_t off = 0;
  auto take = [&](size_t bytes) {
    char* p = ws + off;
    off = (off + bytes + 255) & ~(size_t)255;
    return p;
  };
  bf16* hb      = (bf16*)take((size_t)N_ROWS * H_DIM * 2);
  bf16* extras  = (bf16*)take((size_t)N_ROWS * 32 * 2);
  bf16* z1b     = (bf16*)take((size_t)N_ROWS * H_DIM * 2);
  bf16* W1s     = (bf16*)take((size_t)KCH1 * NSUB_TOT * 512 * 2);
  bf16* W2s     = (bf16*)take((size_t)KCH2 * NSUB_TOT * 512 * 2);
  float* pred_acc = (float*)take((size_t)N_ROWS * 4);
  float* scal     = (float*)take(4 * sizeof(float));

  k_init<<<N_ROWS / 256, 256, 0, stream>>>(pred_acc, scal);
  k_row_prep<<<N_ROWS / 8, 256, 0, stream>>>(h, tg, x0, t, Ws, bs, Wc, bc,
                                             hb, extras, scal);
  k_wprep<<<KCH1 * NSUB_TOT, 256, 0, stream>>>(W1, H_DIM + 3, W1s);
  k_wprep<<<KCH2 * NSUB_TOT, 256, 0, stream>>>(W2, H_DIM, W2s);

  // column-block fastest for L2 reuse of the A row-block
  dim3 gg(H_DIM / 128, N_ROWS / 128);
  k_gemm1<<<gg, 256, 0, stream>>>(hb, extras, W1s, b1, z1b);
  k_gemm2<<<gg, 256, 0, stream>>>(z1b, W2s, b2, W3, pred_acc);

  k_fin_rows<<<N_ROWS / 256, 256, 0, stream>>>(pred_acc, tg, x0, b3, scal);
  k_fin_out<<<1, 32, 0, stream>>>(scal, out);
}
